// ITPLConv_24266565222423
// MI455X (gfx1250) — matrix operand order repaired and checked by reading, not compiled, not run on hardware
//
#include <hip/hip_runtime.h>

typedef __attribute__((ext_vector_type(16))) __bf16 v16bf;
typedef __attribute__((ext_vector_type(8)))  float  v8f;

#define C_IN   128
#define C_OUT  128
#define HW     128
#define TH     4                      // output rows per block
#define TW     32                     // output cols per block
#define SROWS  (TH + 2)               // slab rows (halo)
#define SCOLS  (TW + 2)               // slab cols (halo)
#define PITCH  144                    // LDS c-pitch (bf16); 288B rows, 32B-aligned
#define LDS_ELEMS (SROWS * SCOLS * PITCH)   // 6*34*144 = 29376 ushorts = 57.4 KB

__device__ __forceinline__ unsigned short f2bf(float f) {
    union { float f; unsigned int u; } v; v.f = f;
    unsigned int u = v.u;
    u += 0x7FFFu + ((u >> 16) & 1u);  // round-to-nearest-even
    return (unsigned short)(u >> 16);
}

// ---------------------------------------------------------------------------
// Fold depthwise + pointwise + corr-subtraction into one 3x3 weight, written
// in pre-swizzled A fragment order used here (k = 16*(lane/16) + elem):
//   frag = (tap*4 + kblock)*8 + mtile ; per fragment: lane[0..31] x 16 bf16
// ---------------------------------------------------------------------------
__global__ void fold_weights_kernel(const float* __restrict__ W,
                                    const float* __restrict__ dw,
                                    const float* __restrict__ pw,
                                    unsigned short* __restrict__ wout) {
    int idx = blockIdx.x * blockDim.x + threadIdx.x;
    if (idx >= C_OUT * C_IN * 9) return;
    int t = idx % 9;
    int c = (idx / 9) % C_IN;
    int o = idx / (9 * C_IN);

    float acc = 0.f;
    for (int d = 0; d < C_IN; ++d) {
        float wv = W[(d * C_IN + c) * 9 + t];
        if (t == 0) {                         // corr term hits window tap (0,0)
            float s9 = 0.f;
            #pragma unroll
            for (int tt = 0; tt < 9; ++tt) s9 += W[(d * C_IN + c) * 9 + tt];
            wv -= s9;
        }
        acc += pw[o * C_IN + d] * dw[d] * wv;
    }

    int m  = o >> 4, M = o & 15;
    int kb = c >> 5, k = c & 31;
    int half = k >> 4;                        // lanes 16-31 hold K+16
    int kp   = k - 16 * half;                 // in {0..15}
    int j    = kp >> 1;
    int i    = k & 1;
    int L    = M + 16 * half;
    int frag = (t * 4 + kb) * 8 + m;
    wout[frag * 512 + L * 16 + j * 2 + i] = f2bf(acc);
}

__device__ __forceinline__ void store_ctile(float* __restrict__ p, v8f a) {
    #pragma unroll
    for (int v = 0; v < 8; ++v) p[(size_t)v * HW * HW] = a[v];
}

// ---------------------------------------------------------------------------
// Implicit-GEMM 3x3 conv. Block = 4h x 32w spatial tile x all 128 out chans.
// 8 waves = 4 M-groups (2 M-tiles each) x 2 N-groups (4 N-tiles each).
// Each B fragment (1 ds_load_b128 pair) feeds TWO wmmas (A reuse from L2).
// K loop: 9 taps x 4 c-blocks -> 36 k-steps x 8 wmma = 288 wmma / wave.
// ---------------------------------------------------------------------------
__global__ void __launch_bounds__(256)
conv3x3_wmma_kernel(const float* __restrict__ x,
                    const unsigned short* __restrict__ wfold,
                    float* __restrict__ out) {
    __shared__ unsigned short xs[LDS_ELEMS];

    const int w0 = blockIdx.x * TW;       // 0,32,64,96
    const int h0 = blockIdx.y * TH;       // 0..124
    const int n  = blockIdx.z;            // 0..15

    const int tid  = threadIdx.x;
    const int wave = tid >> 5;
    const int lane = tid & 31;

    // ---- stage x slab: c 0..127, rows h0-1..h0+4, cols w0-1..w0+32, bf16 ----
    for (int g = wave; g < SROWS * C_IN; g += 8) {
        const int c   = g & 127;
        const int dh  = g >> 7;                  // 0..5
        const int row = h0 + dh - 1;
        const bool rok = (row >= 0) && (row < HW);
        const float* src = x + (((size_t)n * C_IN + c) * HW + row) * HW;
        for (int p = lane; p < SCOLS; p += 32) {
            const int col = w0 - 1 + p;
            float v = 0.f;
            if (rok && col >= 0 && col < HW) v = src[col];
            xs[(dh * SCOLS + p) * PITCH + c] = f2bf(v);
        }
    }
    __syncthreads();

    const int mg    = wave & 3;           // M-tiles 2mg, 2mg+1
    const int ng    = wave >> 2;          // N-tiles: rows 2ng,2ng+1 x 2 col-halves
    const int ln15  = lane & 15;
    const int chalf = (lane >> 4) << 4;   // B layout: lanes>=16 hold K+16

    v8f accA[4] = {v8f{}, v8f{}, v8f{}, v8f{}};   // m-tile 2mg
    v8f accB[4] = {v8f{}, v8f{}, v8f{}, v8f{}};   // m-tile 2mg+1

    for (int t = 0; t < 9; ++t) {
        const int kh = t / 3;
        const int kw = t % 3;
        #pragma unroll
        for (int kb = 0; kb < 4; ++kb) {
            // A fragments: contiguous 32B per lane, L2-resident pre-swizzled
            const unsigned short* abase =
                wfold + ((t * 4 + kb) * 8 + 2 * mg) * 512 + lane * 16;
            const v16bf a0 = *(const v16bf*)(abase);
            const v16bf a1 = *(const v16bf*)(abase + 512);

            #pragma unroll
            for (int ni = 0; ni < 4; ++ni) {
                const int r  = 2 * ng + (ni >> 1);      // output row in tile
                const int cb = (ni & 1) * 16;           // col half
                const v16bf b = *(const v16bf*)
                    (xs + ((r + kh) * SCOLS + cb + ln15 + kw) * PITCH
                        + kb * 32 + chalf);
                accA[ni] = __builtin_amdgcn_wmma_f32_16x16x32_bf16(
                               false, a0, false, b, (short)0, accA[ni], false, false);
                accB[ni] = __builtin_amdgcn_wmma_f32_16x16x32_bf16(
                               false, a1, false, b, (short)0, accB[ni], false, false);
            }
        }
    }

    // ---- epilogue: C/D 16x16 f32 layout (VGPR v -> M=v / v+8, N=lane%16) ----
    const int olane = (lane >> 4) << 3;
    #pragma unroll
    for (int ni = 0; ni < 4; ++ni) {
        const int row = h0 + 2 * ng + (ni >> 1);
        const int col = w0 + (ni & 1) * 16 + ln15;
        size_t base = (((size_t)n * C_OUT + 2 * mg * 16 + olane) * HW + row) * HW + col;
        store_ctile(out + base, accA[ni]);
        store_ctile(out + base + (size_t)16 * HW * HW, accB[ni]);
    }
}

extern "C" void kernel_launch(void* const* d_in, const int* in_sizes, int n_in,
                              void* d_out, int out_size, void* d_ws, size_t ws_size,
                              hipStream_t stream) {
    const float* x  = (const float*)d_in[0];   // (16,128,128,128)
    const float* W  = (const float*)d_in[1];   // (128,128,3,3)
    const float* dw = (const float*)d_in[2];   // (128,)
    const float* pw = (const float*)d_in[3];   // (128,128)
    float* out = (float*)d_out;                // (16,128,128,128) f32
    unsigned short* wfold = (unsigned short*)d_ws;  // 288 KB arranged bf16 weights

    // 1) fold + swizzle weights (tiny)
    fold_weights_kernel<<<(C_OUT * C_IN * 9 + 255) / 256, 256, 0, stream>>>(
        W, dw, pw, wfold);

    // 2) implicit-GEMM conv: 4 w-tiles x 32 h-tiles x 16 batch
    dim3 grid(HW / TW, HW / TH, 16);
    conv3x3_wmma_kernel<<<grid, 256, 0, stream>>>(x, wfold, out);
}
